// encoder_44263932952973
// MI455X (gfx1250) — compile-verified, matrix operand order checked by reading, not run on hardware
//
#include <hip/hip_runtime.h>
#include <hip/hip_bf16.h>

// ---------------------------------------------------------------------------
// Types / helpers
// ---------------------------------------------------------------------------
typedef __attribute__((ext_vector_type(16))) __bf16         v16bf;
typedef __attribute__((ext_vector_type(8)))  float          v8f;
typedef __attribute__((ext_vector_type(8)))  unsigned short us8;
typedef __attribute__((ext_vector_type(16))) unsigned short us16;

__device__ inline unsigned short f2us_bf(float f) {
    unsigned u = __builtin_bit_cast(unsigned, f);
    u += 0x7FFFu + ((u >> 16) & 1u);           // round-to-nearest-even
    return (unsigned short)(u >> 16);
}
__device__ inline float us2f_bf(unsigned short s) {
    unsigned u = ((unsigned)s) << 16;
    return __builtin_bit_cast(float, u);
}

#define BN_EPS 1e-5f

// Per-sublayer precomputed params (set up by prep_kernel).
struct SubP {
    const unsigned short* w;   // bf16 weights, padded + swizzled to B-fragment order
    const float* sc;           // folded BN scale  per out channel
    const float* sh;           // folded BN shift  per out channel
};

// ---------------------------------------------------------------------------
// Prep: swizzle W (f32, Cout x CinReal) into bf16 B-fragment order, padded to
// CinP; fold conv bias + BN into per-channel scale/shift.
// Fragment layout: idx = ((nt*(CinP/32)+kc32)*32 + lane)*16 + j
//   o  = nt*16 + (lane&15)
//   c  = kc32*32 + ((j<8)? j : 8+j) + 8*(lane>>4)
// ---------------------------------------------------------------------------
__global__ void prep_kernel(const float* __restrict__ W,  const float* __restrict__ Bi,
                            const float* __restrict__ Ga, const float* __restrict__ Be,
                            const float* __restrict__ Me, const float* __restrict__ Va,
                            unsigned short* __restrict__ wswz,
                            float* __restrict__ scale, float* __restrict__ shift,
                            int CinReal, int CinP, int Cout) {
    const int tid = blockIdx.x * blockDim.x + threadIdx.x;
    const int total = Cout * CinP;
    if (tid < total) {
        int j    = tid & 15;
        int lane = (tid >> 4) & 31;
        int fb   = tid >> 9;                    // nt*(CinP/32) + kc32
        int nk   = CinP >> 5;
        int nt   = fb / nk;
        int kc32 = fb - nt * nk;
        int o    = nt * 16 + (lane & 15);
        int c    = kc32 * 32 + ((j < 8) ? j : (8 + j)) + 8 * (lane >> 4);
        float wv = (c < CinReal) ? W[(size_t)o * CinReal + c] : 0.0f;
        wswz[tid] = f2us_bf(wv);
    }
    if (tid < Cout) {
        float sc = Ga[tid] * rsqrtf(Va[tid] + BN_EPS);
        scale[tid] = sc;
        shift[tid] = sc * (Bi[tid] - Me[tid]) + Be[tid];
    }
}

// ---------------------------------------------------------------------------
// FPS: one workgroup per batch; dist[] lives in LDS; tree argmax per step.
// ---------------------------------------------------------------------------
template<int N, int NPOINT, int T>
__global__ __launch_bounds__(T) void fps_kernel(const float* __restrict__ xyz,
                                                float* __restrict__ new_xyz) {
    __shared__ float dist[N];
    __shared__ float rv[T];
    __shared__ int   ri[T];
    const int b   = blockIdx.x;
    const int tid = threadIdx.x;
    const float* xb = xyz + (size_t)b * N * 3;

    for (int i = tid; i < N; i += T) dist[i] = 1e10f;
    __syncthreads();

    int far = 0;
    for (int s = 0; s < NPOINT; ++s) {
        float cx = xb[far * 3 + 0], cy = xb[far * 3 + 1], cz = xb[far * 3 + 2];
        if (tid == 0) {
            float* nz = new_xyz + ((size_t)b * NPOINT + s) * 3;
            nz[0] = cx; nz[1] = cy; nz[2] = cz;
        }
        float bv = -1.0f; int bi = 0;
        for (int i = tid; i < N; i += T) {
            float dx = xb[i * 3 + 0] - cx;
            float dy = xb[i * 3 + 1] - cy;
            float dz = xb[i * 3 + 2] - cz;
            float d  = dx * dx + dy * dy + dz * dz;
            float dd = fminf(dist[i], d);
            dist[i] = dd;
            if (dd > bv) { bv = dd; bi = i; }
        }
        rv[tid] = bv; ri[tid] = bi;
        __syncthreads();
        for (int st = T >> 1; st > 0; st >>= 1) {
            if (tid < st) {
                if (rv[tid + st] > rv[tid]) { rv[tid] = rv[tid + st]; ri[tid] = ri[tid + st]; }
            }
            __syncthreads();
        }
        far = ri[0];
        __syncthreads();
    }
}

// ---------------------------------------------------------------------------
// Ball query: one thread per centroid; first 32 in-radius indices in index
// order, padded with the first hit.
// ---------------------------------------------------------------------------
__global__ void ballquery_kernel(const float* __restrict__ xyz,
                                 const float* __restrict__ new_xyz,
                                 int* __restrict__ gidx,
                                 int B, int N, int S, float r2) {
    const int t = blockIdx.x * blockDim.x + threadIdx.x;
    if (t >= B * S) return;
    const int b = t / S;
    const float* xb = xyz + (size_t)b * N * 3;
    const float cx = new_xyz[t * 3 + 0];
    const float cy = new_xyz[t * 3 + 1];
    const float cz = new_xyz[t * 3 + 2];
    int* g = gidx + (size_t)t * 32;
    int cnt = 0, first = 0;
    for (int n = 0; n < N && cnt < 32; ++n) {
        float dx = xb[n * 3 + 0] - cx;
        float dy = xb[n * 3 + 1] - cy;
        float dz = xb[n * 3 + 2] - cz;
        float d = dx * dx + dy * dy + dz * dz;
        if (d <= r2) {
            if (cnt == 0) first = n;
            g[cnt++] = n;
        }
    }
    for (int k = cnt; k < 32; ++k) g[k] = first;
}

// ---------------------------------------------------------------------------
// One MLP sublayer on a 32 x CinP bf16 tile in LDS.
// A: two ds_load_b128 per fragment; B: one 32B contiguous load per lane from
// pre-swizzled weights (L2-resident); epilogue: folded BN + ReLU + bf16 pack.
// ---------------------------------------------------------------------------
template<int WAVES>
__device__ inline void mlp_sublayer(const unsigned short* __restrict__ inBuf,
                                    unsigned short* __restrict__ outBuf,
                                    int CinP, int Cout, SubP P) {
    const int lane = threadIdx.x & 31;
    const int wave = threadIdx.x >> 5;
    const int lm   = lane & 15;
    const int mh   = lane >> 4;
    const int nk   = CinP >> 5;
    const int tiles = 2 * (Cout >> 4);

    for (int t = wave; t < tiles; t += WAVES) {
        const int mt = t & 1;
        const int nt = t >> 1;
        const int o  = nt * 16 + lm;
        const int m  = mt * 16 + lm;
        const unsigned short* wp = P.w + (size_t)nt * nk * 512 + lane * 16;
        const unsigned short* ap = inBuf + (size_t)m * CinP + 8 * mh;
        v8f acc = {};
        for (int kc32 = 0; kc32 < nk; ++kc32) {
            us8 alo = *(const us8*)(ap + kc32 * 32);
            us8 ahi = *(const us8*)(ap + kc32 * 32 + 16);
            us16 av = __builtin_shufflevector(alo, ahi,
                      0, 1, 2, 3, 4, 5, 6, 7, 8, 9, 10, 11, 12, 13, 14, 15);
            v16bf a = __builtin_bit_cast(v16bf, av);
            us16 bv = *(const us16*)(wp + (size_t)kc32 * 512);
            v16bf bm = __builtin_bit_cast(v16bf, bv);
            acc = __builtin_amdgcn_wmma_f32_16x16x32_bf16(false, a, false, bm,
                                                          (short)0, acc, false, false);
        }
        const float sc = P.sc[o];
        const float sh = P.sh[o];
#pragma unroll
        for (int r = 0; r < 8; ++r) {
            int mrow = mt * 16 + r + 8 * mh;                  // C/D layout row
            float val = fmaxf(0.0f, acc[r] * sc + sh);
            outBuf[(size_t)mrow * Cout + o] = f2us_bf(val);
        }
    }
}

// ---------------------------------------------------------------------------
// Fused group + 3-layer MLP + max-over-K. One workgroup per (b, s) group.
// ---------------------------------------------------------------------------
template<int N, int S, int C_PTS, int CINP, int C0, int C1, int C2, int CBUF, int THREADS>
__global__ __launch_bounds__(THREADS) void group_mlp_kernel(
    const float* __restrict__ xyz, const float* __restrict__ pts,
    const float* __restrict__ new_xyz, const int* __restrict__ gidx,
    SubP P0, SubP P1, SubP P2,
    float* __restrict__ pts_out) {
    constexpr int WAVES = THREADS / 32;
    __shared__ __align__(16) unsigned short bufA[32 * CBUF];
    __shared__ __align__(16) unsigned short bufB[32 * CBUF];
    const int bs  = blockIdx.x;
    const int b   = bs / S;
    const int tid = threadIdx.x;
    const int* gi = gidx + (size_t)bs * 32;
    const float* nz = new_xyz + (size_t)bs * 3;
    const float cx = nz[0], cy = nz[1], cz = nz[2];

    // Gather: feat[k, c] = [xyz(rel) | pts], zero-pad to CINP, store bf16.
    for (int e = tid; e < 32 * CINP; e += THREADS) {
        int k = e / CINP, c = e % CINP;
        int n = gi[k];
        float v = 0.0f;
        if (c < 3)              v = xyz[((size_t)b * N + n) * 3 + c] -
                                    (c == 0 ? cx : (c == 1 ? cy : cz));
        else if (c < 3 + C_PTS) v = pts[((size_t)b * N + n) * C_PTS + (c - 3)];
        bufA[(size_t)k * CINP + c] = f2us_bf(v);
    }
    __syncthreads();

    mlp_sublayer<WAVES>(bufA, bufB, CINP, C0, P0); __syncthreads();
    mlp_sublayer<WAVES>(bufB, bufA, C0,   C1, P1); __syncthreads();
    mlp_sublayer<WAVES>(bufA, bufB, C1,   C2, P2); __syncthreads();

    // Max over the 32 neighbors (post-ReLU => >= 0).
    for (int o = tid; o < C2; o += THREADS) {
        float mx = 0.0f;
#pragma unroll 4
        for (int k = 0; k < 32; ++k) mx = fmaxf(mx, us2f_bf(bufB[(size_t)k * C2 + o]));
        pts_out[(size_t)bs * C2 + o] = mx;
    }
}

// ---------------------------------------------------------------------------
// Final projection: out[b, 0, z] = sum_s pts[b, s, z] * w[s] + bias
// ---------------------------------------------------------------------------
__global__ void conv0_kernel(const float* __restrict__ pts,  // (B, 16, 512)
                             const float* __restrict__ w,    // (16,)
                             const float* __restrict__ bias, // (1,)
                             float* __restrict__ out) {      // (B, 1, 512)
    const int b = blockIdx.x;
    const int z = threadIdx.x;
    float acc = bias[0];
#pragma unroll
    for (int s = 0; s < 16; ++s)
        acc += pts[((size_t)b * 16 + s) * 512 + z] * w[s];
    out[(size_t)b * 512 + z] = acc;
}

// ---------------------------------------------------------------------------
// Host launcher
// ---------------------------------------------------------------------------
extern "C" void kernel_launch(void* const* d_in, const int* in_sizes, int n_in,
                              void* d_out, int out_size, void* d_ws, size_t ws_size,
                              hipStream_t stream) {
    (void)in_sizes; (void)n_in; (void)out_size; (void)ws_size;
    const float* xyz = (const float*)d_in[0];
    const float* c0w = (const float*)d_in[73];
    const float* c0b = (const float*)d_in[74];
    float* out = (float*)d_out;

    char* ws = (char*)d_ws;
    size_t off = 0;
    auto alloc = [&](size_t bytes) -> void* {
        off = (off + 255) & ~(size_t)255;
        void* p = ws + off;
        off += bytes;
        return p;
    };
    float* nx1 = (float*)alloc(8ull * 1024 * 3 * 4);
    float* p1  = (float*)alloc(8ull * 1024 * 64 * 4);
    float* nx2 = (float*)alloc(8ull * 256 * 3 * 4);
    float* p2  = (float*)alloc(8ull * 256 * 128 * 4);
    float* nx3 = (float*)alloc(8ull * 64 * 3 * 4);
    float* p3  = (float*)alloc(8ull * 64 * 256 * 4);
    float* nx4 = (float*)alloc(8ull * 16 * 3 * 4);
    float* p4  = (float*)alloc(8ull * 16 * 512 * 4);
    int*  gidx = (int*)alloc(8ull * 1024 * 32 * 4);

    // Sublayer dims: {CinReal, CinP, Cout} x 3 per stage.
    static const int DIMS[4][3][3] = {
        { {  6,  32,  32}, { 32,  32,  32}, { 32,  32,  64} },
        { { 67,  96,  64}, { 64,  64,  64}, { 64,  64, 128} },
        { {131, 160, 128}, {128, 128, 128}, {128, 128, 256} },
        { {259, 288, 256}, {256, 256, 256}, {256, 256, 512} },
    };

    SubP SP[4][3];
    for (int L = 0; L < 4; ++L) {
        for (int s = 0; s < 3; ++s) {
            const int CinP = DIMS[L][s][1], Cout = DIMS[L][s][2];
            unsigned short* wz = (unsigned short*)alloc((size_t)Cout * CinP * 2);
            float* sc = (float*)alloc((size_t)Cout * 4);
            float* sh = (float*)alloc((size_t)Cout * 4);
            SP[L][s] = SubP{ wz, sc, sh };
        }
    }

    // Prep: swizzle weights to bf16 fragments + fold BN (inputs: per-sublayer
    // {w,b,gamma,beta,mean,var} at d_in[1 + (L*3+s)*6 + k]).
    for (int L = 0; L < 4; ++L) {
        for (int s = 0; s < 3; ++s) {
            const int base = 1 + (L * 3 + s) * 6;
            const int CinReal = DIMS[L][s][0], CinP = DIMS[L][s][1], Cout = DIMS[L][s][2];
            const int total = Cout * CinP;
            prep_kernel<<<(total + 255) / 256, 256, 0, stream>>>(
                (const float*)d_in[base + 0], (const float*)d_in[base + 1],
                (const float*)d_in[base + 2], (const float*)d_in[base + 3],
                (const float*)d_in[base + 4], (const float*)d_in[base + 5],
                (unsigned short*)SP[L][s].w, (float*)SP[L][s].sc, (float*)SP[L][s].sh,
                CinReal, CinP, Cout);
        }
    }

    const int B = 8;

    // ---- SA stage 1: N=8192 -> S=1024, r=0.1, MLP 6->32->32->64 ----
    fps_kernel<8192, 1024, 1024><<<B, 1024, 0, stream>>>(xyz, nx1);
    ballquery_kernel<<<(B * 1024 + 255) / 256, 256, 0, stream>>>(
        xyz, nx1, gidx, B, 8192, 1024, 0.01f);
    group_mlp_kernel<8192, 1024, 3, 32, 32, 32, 64, 64, 128>
        <<<B * 1024, 128, 0, stream>>>(xyz, xyz, nx1, gidx,
                                       SP[0][0], SP[0][1], SP[0][2], p1);

    // ---- SA stage 2: N=1024 -> S=256, r=0.2, MLP 67->64->64->128 ----
    fps_kernel<1024, 256, 1024><<<B, 1024, 0, stream>>>(nx1, nx2);
    ballquery_kernel<<<(B * 256 + 255) / 256, 256, 0, stream>>>(
        nx1, nx2, gidx, B, 1024, 256, 0.04f);
    group_mlp_kernel<1024, 256, 64, 96, 64, 64, 128, 128, 128>
        <<<B * 256, 128, 0, stream>>>(nx1, p1, nx2, gidx,
                                      SP[1][0], SP[1][1], SP[1][2], p2);

    // ---- SA stage 3: N=256 -> S=64, r=0.4, MLP 131->128->128->256 ----
    fps_kernel<256, 64, 256><<<B, 256, 0, stream>>>(nx2, nx3);
    ballquery_kernel<<<(B * 64 + 255) / 256, 256, 0, stream>>>(
        nx2, nx3, gidx, B, 256, 64, 0.16f);
    group_mlp_kernel<256, 64, 128, 160, 128, 128, 256, 256, 256>
        <<<B * 64, 256, 0, stream>>>(nx2, p2, nx3, gidx,
                                     SP[2][0], SP[2][1], SP[2][2], p3);

    // ---- SA stage 4: N=64 -> S=16, r=0.8, MLP 259->256->256->512 ----
    fps_kernel<64, 16, 64><<<B, 64, 0, stream>>>(nx3, nx4);
    ballquery_kernel<<<(B * 16 + 255) / 256, 256, 0, stream>>>(
        nx3, nx4, gidx, B, 64, 16, 0.64f);
    group_mlp_kernel<64, 16, 256, 288, 256, 256, 512, 512, 256>
        <<<B * 16, 256, 0, stream>>>(nx3, p3, nx4, gidx,
                                     SP[3][0], SP[3][1], SP[3][2], p4);

    // ---- Final projection (8,16,512) -> (8,1,512) ----
    conv0_kernel<<<B, 512, 0, stream>>>(p4, c0w, c0b, out);
}
